// FBSNN_76948634075302
// MI455X (gfx1250) — matrix operand order repaired and checked by reading, not compiled, not run on hardware
//
#include <hip/hip_runtime.h>
#include <hip/hip_fp16.h>

typedef __attribute__((ext_vector_type(16))) _Float16 v16h;
typedef __attribute__((ext_vector_type(8)))  _Float16 v8h;
typedef __attribute__((ext_vector_type(8)))  float    v8f;

#define R_       0.05f
#define MU_      0.05f
#define SIGMA_   0.4f
#define KSTRIKE_ 100.0f
#define DT_      0.02f
#define LAM1_    1.0f
#define LAM2_    1.0f
#define LAM3_    1.0f

#define MPATH   1024
#define NSTEP   50
#define DDIM    100
#define NT      51
#define ROWS_MAIN (MPATH*NT)     // 52224
#define ROWS_PAD  52352          // 409*128 >= 52224 + 50 zero-input rows
#define K0      128              // padded input width
#define H1      1024
#define H2      512
#define H3      256

// ---------------------------------------------------------------------------
// Fragment-native layouts (CDNA5 ISA 7.12.2, wave32).
// A-matrix 16x32 f16 fragment: lane = (row&15) + 16*lh, lh=(k'>>3)&1,
//                              e = (k'&7) + 8*(k'>>4),  k'=k&31.
// B-matrix 32x16 f16 fragment: lane = (col&15) + 16*lh, same lh and e as A.
// Fragments stored as 512 contiguous halves at offset lane*16 + e.
// A frags ordered row-block-major (K contiguous); B frags col-block-major.
// ---------------------------------------------------------------------------

__device__ __forceinline__ size_t afrag(long row, int k, int Kdim)
{
    int kp = k & 31;
    int lh = (kp >> 3) & 1;
    int e  = (kp & 7) + ((kp >> 4) << 3);
    return ((size_t)((row >> 4) * (Kdim >> 5) + (k >> 5)) << 9)
         + ((size_t)(((int)(row & 15)) + (lh << 4)) << 4) + e;
}

__device__ __forceinline__ size_t bfrag(int k, int col, int Kdim)
{
    int kp = k & 31;
    int lh = (kp >> 3) & 1;
    int e  = (kp & 7) + ((kp >> 4) << 3);
    return ((size_t)((col >> 4) * (Kdim >> 5) + (k >> 5)) << 9)
         + ((size_t)((col & 15) + (lh << 4)) << 4) + e;
}

// ---------------------------------------------------------------------------
// GEMM: A[R x K] (A-frag layout) @ B[K x N] (B-frag layout) -> C[R x N]
// stored in A-frag layout (ready to be the next layer's A operand).
// Block tile 128x64: 8 waves, each wave 16 rows x 64 cols (4 accumulators).
// Register-double-buffered LDS staging; all LDS traffic is b128.
// mode 0: C = relu(acc + bias[col]) ; mode 1: C = acc * (C_prev > 0) in place.
// ---------------------------------------------------------------------------
__global__ __launch_bounds__(256)
void gemm_wmma(const _Float16* __restrict__ A,
               const _Float16* __restrict__ B,
               _Float16* __restrict__ C,
               const float* __restrict__ bias,
               int K, int Nout, int mode)
{
    __shared__ alignas(32) _Float16 sA[8 * 512];   // 8 row fragments
    __shared__ alignas(32) _Float16 sB[4 * 512];   // 4 col fragments
    const int tid  = threadIdx.x;
    const int lane = tid & 31;
    const int wave = tid >> 5;
    const int Kf   = K >> 5;
    const long rowFragBase = (long)blockIdx.y * 8;
    const int  colFragBase = blockIdx.x * 4;

    const _Float16* pa = A + ((size_t)(rowFragBase + (tid >> 5)) * Kf << 9) + (size_t)(tid & 31) * 16;
    const _Float16* pb = B + ((size_t)(colFragBase + (tid >> 6)) * Kf << 9) + (size_t)(tid & 63) * 8;

    v8f acc0 = {}, acc1 = {}, acc2 = {}, acc3 = {};
    v16h ra = *(const v16h*)pa;
    v8h  rb = *(const v8h*)pb;

    for (int kb = 0; kb < Kf; ++kb) {
        *(v16h*)&sA[tid * 16] = ra;
        *(v8h*) &sB[tid * 8]  = rb;
        __syncthreads();
        if (kb + 1 < Kf) {                       // prefetch next K-step
            ra = *(const v16h*)(pa + ((size_t)(kb + 1) << 9));
            rb = *(const v8h*) (pb + ((size_t)(kb + 1) << 9));
        }
        v16h a  = *(const v16h*)&sA[wave * 512 + lane * 16];
        v16h b0 = *(const v16h*)&sB[0 * 512 + lane * 16];
        v16h b1 = *(const v16h*)&sB[1 * 512 + lane * 16];
        v16h b2 = *(const v16h*)&sB[2 * 512 + lane * 16];
        v16h b3 = *(const v16h*)&sB[3 * 512 + lane * 16];
        acc0 = __builtin_amdgcn_wmma_f32_16x16x32_f16(false, a, false, b0, (short)0, acc0, false, false);
        acc1 = __builtin_amdgcn_wmma_f32_16x16x32_f16(false, a, false, b1, (short)0, acc1, false, false);
        acc2 = __builtin_amdgcn_wmma_f32_16x16x32_f16(false, a, false, b2, (short)0, acc2, false, false);
        acc3 = __builtin_amdgcn_wmma_f32_16x16x32_f16(false, a, false, b3, (short)0, acc3, false, false);
        __syncthreads();
    }

    // epilogue: D layout (vgpr r -> rows r, r+8; col = lane&15) -> A-frag store
    const int lh = lane >> 4;
    const long rowB = (long)blockIdx.y * 128 + wave * 16 + (lh << 3);
    const int  colB = blockIdx.x * 64 + (lane & 15);
    v8f* accs[4] = { &acc0, &acc1, &acc2, &acc3 };
    #pragma unroll
    for (int f = 0; f < 4; ++f) {
        int col = colB + f * 16;
        #pragma unroll
        for (int r = 0; r < 8; ++r) {
            long  row = rowB + r;
            size_t idx = afrag(row, col, Nout);
            float v = (*accs[f])[r];
            if (mode == 0) {
                v += bias[col];
                C[idx] = (_Float16)(v > 0.f ? v : 0.f);
            } else {
                float m = ((float)C[idx] > 0.f) ? 1.f : 0.f;
                C[idx] = (_Float16)(v * m);
            }
        }
    }
}

// Final backward GEMM: G1[R x 1024] @ W1T[1024 x 128] -> DuDx f32 (drop col 0)
__global__ __launch_bounds__(256)
void gemm_wmma_dx(const _Float16* __restrict__ A,
                  const _Float16* __restrict__ B,
                  float* __restrict__ DU, int K)
{
    __shared__ alignas(32) _Float16 sA[8 * 512];
    __shared__ alignas(32) _Float16 sB[4 * 512];
    const int tid  = threadIdx.x;
    const int lane = tid & 31;
    const int wave = tid >> 5;
    const int Kf   = K >> 5;
    const long rowFragBase = (long)blockIdx.y * 8;
    const int  colFragBase = blockIdx.x * 4;

    const _Float16* pa = A + ((size_t)(rowFragBase + (tid >> 5)) * Kf << 9) + (size_t)(tid & 31) * 16;
    const _Float16* pb = B + ((size_t)(colFragBase + (tid >> 6)) * Kf << 9) + (size_t)(tid & 63) * 8;

    v8f acc0 = {}, acc1 = {}, acc2 = {}, acc3 = {};
    v16h ra = *(const v16h*)pa;
    v8h  rb = *(const v8h*)pb;

    for (int kb = 0; kb < Kf; ++kb) {
        *(v16h*)&sA[tid * 16] = ra;
        *(v8h*) &sB[tid * 8]  = rb;
        __syncthreads();
        if (kb + 1 < Kf) {
            ra = *(const v16h*)(pa + ((size_t)(kb + 1) << 9));
            rb = *(const v8h*) (pb + ((size_t)(kb + 1) << 9));
        }
        v16h a  = *(const v16h*)&sA[wave * 512 + lane * 16];
        v16h b0 = *(const v16h*)&sB[0 * 512 + lane * 16];
        v16h b1 = *(const v16h*)&sB[1 * 512 + lane * 16];
        v16h b2 = *(const v16h*)&sB[2 * 512 + lane * 16];
        v16h b3 = *(const v16h*)&sB[3 * 512 + lane * 16];
        acc0 = __builtin_amdgcn_wmma_f32_16x16x32_f16(false, a, false, b0, (short)0, acc0, false, false);
        acc1 = __builtin_amdgcn_wmma_f32_16x16x32_f16(false, a, false, b1, (short)0, acc1, false, false);
        acc2 = __builtin_amdgcn_wmma_f32_16x16x32_f16(false, a, false, b2, (short)0, acc2, false, false);
        acc3 = __builtin_amdgcn_wmma_f32_16x16x32_f16(false, a, false, b3, (short)0, acc3, false, false);
        __syncthreads();
    }

    const int lh = lane >> 4;
    const long rowB = (long)blockIdx.y * 128 + wave * 16 + (lh << 3);
    const int  colB = blockIdx.x * 64 + (lane & 15);
    v8f* accs[4] = { &acc0, &acc1, &acc2, &acc3 };
    #pragma unroll
    for (int f = 0; f < 4; ++f) {
        int col = colB + f * 16;
        if (col >= 1 && col <= DDIM) {
            #pragma unroll
            for (int r = 0; r < 8; ++r)
                DU[(rowB + r) * DDIM + col - 1] = (*accs[f])[r];
        }
    }
}

// --------------------------- prep / glue kernels ---------------------------

__global__ void prep_w1(const float* W1, _Float16* W1p, _Float16* W1T)
{
    int i = blockIdx.x * blockDim.x + threadIdx.x;
    if (i >= K0 * H1) return;
    int r = i / H1, c = i % H1;
    _Float16 v = (r < DDIM + 1) ? (_Float16)W1[r * H1 + c] : (_Float16)0.f;
    W1p[bfrag(r, c, K0)] = v;        // forward:  B of [128 x 1024]
    W1T[bfrag(c, r, H1)] = v;        // backward: B of [1024 x 128]
}

__global__ void prep_w2(const float* W2, _Float16* W2h, _Float16* W2T)
{
    int i = blockIdx.x * blockDim.x + threadIdx.x;
    if (i >= H1 * H2) return;
    int r = i / H2, c = i % H2;
    _Float16 v = (_Float16)W2[i];
    W2h[bfrag(r, c, H1)] = v;
    W2T[bfrag(c, r, H2)] = v;
}

__global__ void prep_w3(const float* W3, _Float16* W3h, _Float16* W3T)
{
    int i = blockIdx.x * blockDim.x + threadIdx.x;
    if (i >= H2 * H3) return;
    int r = i / H3, c = i % H3;
    _Float16 v = (_Float16)W3[i];
    W3h[bfrag(r, c, H2)] = v;
    W3T[bfrag(c, r, H3)] = v;
}

// fill H0 (A-frag layout of [ROWS_PAD x 128]): time column + zeros elsewhere
__global__ void init_h0(_Float16* H0, const float* t)
{
    long i = (long)blockIdx.x * blockDim.x + threadIdx.x;
    if (i >= (long)ROWS_PAD * K0) return;
    long frag = i >> 9;
    int  w = (int)(i & 511), lane = w >> 4, e = w & 15;
    int  kb = (int)(frag & 3);                        // K0/32 = 4
    long row = ((frag >> 2) << 4) + (lane & 15);
    int  k = kb * 32 + (e & 7) + ((lane >> 4) << 3) + ((e >> 3) << 4);
    float v = 0.f;
    if (k == 0) {
        if (row < ROWS_MAIN) {
            int n = (int)(row / MPATH), m = (int)(row % MPATH);
            v = t[(long)m * NT + n];
        } else if (row < ROWS_MAIN + NSTEP) {
            v = t[(int)(row - ROWS_MAIN) + 1];        // t[0][j+1][0]
        }
    }
    H0[i] = (_Float16)v;
}

// GBM scan: writes f32 X_full (d_out) + f16 H0 X-columns (A-frag layout)
__global__ void path_scan(const float* __restrict__ W, const float* __restrict__ Xi,
                          float* __restrict__ Xout, _Float16* __restrict__ H0)
{
    int i = blockIdx.x * blockDim.x + threadIdx.x;
    if (i >= MPATH * DDIM) return;
    int m = i / DDIM, d = i % DDIM;
    float x = Xi[(long)m * DDIM + d];
    Xout[(long)m * NT * DDIM + d] = x;
    H0[afrag(m, 1 + d, K0)] = (_Float16)x;
    const float drift = (MU_ - 0.5f * SIGMA_ * SIGMA_) * DT_;
    float wprev = W[((long)m * NT) * DDIM + d];
    for (int n = 1; n <= NSTEP; ++n) {
        float wcur = W[((long)m * NT + n) * DDIM + d];
        x = x * __expf(drift + SIGMA_ * (wcur - wprev));
        wprev = wcur;
        Xout[(long)m * NT * DDIM + (long)n * DDIM + d] = x;
        H0[afrag((long)n * MPATH + m, 1 + d, K0)] = (_Float16)x;
    }
}

// u = A3 @ Wo + bo ; A3 in A-frag layout (wave per 16-row fragment block)
__global__ __launch_bounds__(256)
void out_layer(const _Float16* __restrict__ A3, const float* __restrict__ Wo,
               const float* __restrict__ bo, float* __restrict__ U)
{
    __shared__ float sWo[H3];
    int tid = threadIdx.x;
    sWo[tid] = Wo[tid];               // 256 threads == H3
    __syncthreads();
    int wave = tid >> 5, lane = tid & 31;
    long rb = (long)blockIdx.x * 8 + wave;
    const int Kf = H3 >> 5;           // 8
    int lh = lane >> 4;
    float s = 0.f;
    for (int kb = 0; kb < Kf; ++kb) {
        v16h v = *(const v16h*)(A3 + ((size_t)rb * Kf + kb) * 512 + (size_t)lane * 16);
        #pragma unroll
        for (int e = 0; e < 16; ++e) {
            int k = kb * 32 + (e & 7) + (lh << 3) + ((e >> 3) << 4);
            s += (float)v[e] * sWo[k];
        }
    }
    s += __shfl_xor(s, 16, 32);       // lanes l and l+16 hold the same row
    if (lane < 16) U[rb * 16 + lane] = s + bo[0];
}

// backward seed: G3 = mask(A3) * Wo[k]   (in place, A-frag layout, K=256)
__global__ void bwd3(_Float16* A3, const float* Wo, long n)
{
    long i = (long)blockIdx.x * blockDim.x + threadIdx.x;
    if (i >= n) return;
    int w = (int)(i & 511), lane = w >> 4, e = w & 15;
    int kb = (int)((i >> 9) & 7);                     // H3/32 = 8
    int k = kb * 32 + (e & 7) + ((lane >> 4) << 3) + ((e >> 3) << 4);
    float m = ((float)A3[i] > 0.f) ? 1.f : 0.f;
    A3[i] = (_Float16)(Wo[k] * m);
}

__global__ void write_y(const float* __restrict__ U, float* __restrict__ Yout,
                        float* __restrict__ Yscalar)
{
    int i = blockIdx.x * blockDim.x + threadIdx.x;
    if (i >= ROWS_MAIN) return;
    int m = i / NT, n = i % NT;
    Yout[i] = U[(long)n * MPATH + m];
    if (i == 0) Yscalar[0] = U[0];
}

__global__ __launch_bounds__(1024)
void loss_kernel(const float* __restrict__ U, const float* __restrict__ DU,
                 const float* __restrict__ Xout, const float* __restrict__ Wmat,
                 float* __restrict__ out_loss)
{
    __shared__ float red[1024];
    int m = threadIdx.x;
    float acc = 0.f;
    for (int n = 0; n < NSTEP; ++n) {
        float Y0 = U[(long)n * MPATH + m];
        float Y1 = U[(long)(n + 1) * MPATH + m];
        float a  = Y0 + R_ * Y0 * DT_;
        const float* Xr  = Xout + (long)m * NT * DDIM + (long)n * DDIM;
        const float* Dur = DU + ((long)n * MPATH + m) * DDIM;
        const float* W0r = Wmat + ((long)m * NT + n) * DDIM;
        const float* W1r = W0r + DDIM;
        for (int d = 0; d < DDIM; ++d) {
            float X0 = Xr[d], Du = Dur[d];
            float dW = W1r[d] - W0r[d];
            float yt = a + (MU_ - R_) * X0 * Du + SIGMA_ * Du * X0 * dW;
            float df = Y1 - yt;
            acc += df * df;
        }
    }
    if (m < NSTEP) {
        float y = U[ROWS_MAIN + m];
        acc += LAM3_ * (float)MPATH * y * y;
    }
    {
        const float* XN = Xout + (long)m * NT * DDIM + (long)NSTEP * DDIM;
        float mx = XN[0]; int am = 0;
        for (int d = 1; d < DDIM; ++d) { float v = XN[d]; if (v > mx) { mx = v; am = d; } }
        float g  = mx - KSTRIKE_;
        float gr = (g > 0.f) ? 1.f : 0.f;
        g = (g > 0.f) ? g : 0.f;
        float YN = U[(long)NSTEP * MPATH + m];
        float df = YN - g;
        acc += LAM1_ * df * df;
        const float* DuN = DU + ((long)NSTEP * MPATH + m) * DDIM;
        for (int d = 0; d < DDIM; ++d) {
            float Dg = (d == am) ? gr : 0.f;
            float e  = DuN[d] - Dg;
            acc += LAM2_ * e * e;
        }
    }
    red[m] = acc;
    __syncthreads();
    for (int s = 512; s > 0; s >>= 1) {
        if (m < s) red[m] += red[m + s];
        __syncthreads();
    }
    if (m == 0)
        out_loss[0] = red[0] / ((float)NSTEP * (1.f + LAM3_) + LAM1_ + LAM2_);
}

// ---------------------------------------------------------------------------

extern "C" void kernel_launch(void* const* d_in, const int* in_sizes, int n_in,
                              void* d_out, int out_size, void* d_ws, size_t ws_size,
                              hipStream_t stream)
{
    (void)in_sizes; (void)n_in; (void)out_size; (void)ws_size;
    const float* t_in = (const float*)d_in[0];
    const float* W_in = (const float*)d_in[1];
    const float* Xi   = (const float*)d_in[2];
    const float* W1   = (const float*)d_in[3];
    const float* b1   = (const float*)d_in[4];
    const float* W2   = (const float*)d_in[5];
    const float* b2   = (const float*)d_in[6];
    const float* W3   = (const float*)d_in[7];
    const float* b3   = (const float*)d_in[8];
    const float* Wo   = (const float*)d_in[9];
    const float* bo   = (const float*)d_in[10];

    float* out = (float*)d_out;
    const long XSZ = (long)MPATH * NT * DDIM;
    const long YSZ = (long)MPATH * NT;
    float* out_loss = out;
    float* out_X    = out + 1;
    float* out_Y    = out + 1 + XSZ;
    float* out_sc   = out + 1 + XSZ + YSZ;

    char* ws = (char*)d_ws;
    size_t off = 0;
    auto take = [&](size_t bytes) -> char* {
        char* p = ws + off;
        off = (off + bytes + 255) & ~(size_t)255;
        return p;
    };
    _Float16* H0  = (_Float16*)take((size_t)ROWS_PAD * K0 * 2);
    _Float16* A1  = (_Float16*)take((size_t)ROWS_PAD * H1 * 2);
    _Float16* A2  = (_Float16*)take((size_t)ROWS_PAD * H2 * 2);
    _Float16* A3  = (_Float16*)take((size_t)ROWS_PAD * H3 * 2);
    float*    U   = (float*)   take((size_t)ROWS_PAD * 4);
    float*    DU  = (float*)   take((size_t)ROWS_PAD * DDIM * 4);
    _Float16* W1p = (_Float16*)take((size_t)K0 * H1 * 2);
    _Float16* W1T = (_Float16*)take((size_t)H1 * K0 * 2);
    _Float16* W2h = (_Float16*)take((size_t)H1 * H2 * 2);
    _Float16* W2T = (_Float16*)take((size_t)H2 * H1 * 2);
    _Float16* W3h = (_Float16*)take((size_t)H2 * H3 * 2);
    _Float16* W3T = (_Float16*)take((size_t)H3 * H2 * 2);

    const int GM = ROWS_PAD / 128;   // 409 row tiles

    prep_w1<<<(K0 * H1 + 255) / 256, 256, 0, stream>>>(W1, W1p, W1T);
    prep_w2<<<(H1 * H2 + 255) / 256, 256, 0, stream>>>(W2, W2h, W2T);
    prep_w3<<<(H2 * H3 + 255) / 256, 256, 0, stream>>>(W3, W3h, W3T);

    init_h0<<<(int)(((long)ROWS_PAD * K0 + 255) / 256), 256, 0, stream>>>(H0, t_in);
    path_scan<<<(MPATH * DDIM + 255) / 256, 256, 0, stream>>>(W_in, Xi, out_X, H0);

    // forward MLP
    gemm_wmma<<<dim3(H1 / 64, GM), 256, 0, stream>>>(H0, W1p, A1, b1, K0, H1, 0);
    gemm_wmma<<<dim3(H2 / 64, GM), 256, 0, stream>>>(A1, W2h, A2, b2, H1, H2, 0);
    gemm_wmma<<<dim3(H3 / 64, GM), 256, 0, stream>>>(A2, W3h, A3, b3, H2, H3, 0);
    out_layer<<<GM, 256, 0, stream>>>(A3, Wo, bo, U);

    // backward VJP (gradients overwrite activations in place)
    bwd3<<<(int)(((long)ROWS_PAD * H3 + 255) / 256), 256, 0, stream>>>(A3, Wo,
                                                       (long)ROWS_PAD * H3);
    gemm_wmma<<<dim3(H2 / 64, GM), 256, 0, stream>>>(A3, W3T, A2, nullptr, H3, H2, 1);
    gemm_wmma<<<dim3(H1 / 64, GM), 256, 0, stream>>>(A2, W2T, A1, nullptr, H2, H1, 1);
    gemm_wmma_dx<<<dim3(K0 / 64, GM), 256, 0, stream>>>(A1, W1T, DU, H1);

    write_y<<<(ROWS_MAIN + 255) / 256, 256, 0, stream>>>(U, out_Y, out_sc);
    loss_kernel<<<1, 1024, 0, stream>>>(U, DU, out_X, W_in, out_loss);
}
